// GroupGSDConv_7301444403498
// MI455X (gfx1250) — matrix operand order repaired and checked by reading, not compiled, not run on hardware
//
#include <hip/hip_runtime.h>

// GSD conv as implicit GEMM on v_wmma_f32_16x16x32_bf16 with error-compensated
// bf16 split (hi+lo): out ~= Whi*Xhi + Whi*Xlo + Wlo*Xhi  (~fp32 accuracy).
//
// Roofline: 19.3 GFLOP vs 128 MiB HBM (~5.5us @ 23.3 TB/s). bf16 WMMA x3
// passes ~= memory-balanced; f32 WMMA (16x16x4) would be ~3x slower.
//
// v2: zero halo columns in LDS (8 each side) -> unconditional B-fragment
// ds_load_b128 (no exec-mask juggling in the 108-WMMA inner pipeline).

typedef __attribute__((ext_vector_type(16))) __bf16 v16bf;
typedef __attribute__((ext_vector_type(8)))  float  v8f;

#define BATCH 16
#define CH    64
#define HW    128
#define KTOT  576            // 9 taps * 64 ic
#define NDIL  6
#define IC_PAD 72            // 144B px stride: 16B aligned, conflict-free b128
#define HALO  8              // max |shift| = 6, round up to 8
#define PXP   (HW + 2 * HALO)            // 144 padded pixel columns
#define WPACK_PER_D (64 * KTOT)

union Frag {
  v16bf v;
  uint4 q[2];
};

__device__ __forceinline__ void split_bf16(float x, __bf16& hi, __bf16& lo) {
  hi = (__bf16)x;                 // RNE
  lo = (__bf16)(x - (float)hi);   // residual captures ~8 more mantissa bits
}

// ---------------------------------------------------------------------------
// Kernel 1: per-dilation channel lists + weight repack to A-fragment layout.
// k ordering: k = (kh*3+kw)*64 + ic, contiguous per oc-slot -> A frag is two
// aligned b128 loads. Slots >= count are zero (padded GEMM rows).
// ---------------------------------------------------------------------------
__global__ void gsd_pack_weights(const float* __restrict__ w,
                                 const int* __restrict__ offs,
                                 int* __restrict__ counts,
                                 int* __restrict__ chan,
                                 __bf16* __restrict__ whi,
                                 __bf16* __restrict__ wlo) {
  const int d = blockIdx.x;            // dilation = d+1
  __shared__ int s_cnt;
  __shared__ int s_chan[CH];
  if (threadIdx.x == 0) {
    int c = 0;
    for (int i = 0; i < CH; ++i)
      if (offs[i] == d + 1) s_chan[c++] = i;
    s_cnt = c;
    counts[d] = c;
    for (int i = c; i < CH; ++i) s_chan[i] = 0;
    for (int i = 0; i < CH; ++i) chan[d * CH + i] = s_chan[i];
  }
  __syncthreads();
  const int cnt = s_cnt;
  for (int i = threadIdx.x; i < WPACK_PER_D; i += blockDim.x) {
    const int slot = i / KTOT;
    const int k = i % KTOT;
    float v = 0.0f;
    if (slot < cnt) {
      const int oc  = s_chan[slot];
      const int ic  = k & 63;
      const int tap = k >> 6;          // kh*3+kw
      const int kh = tap / 3, kw = tap % 3;
      v = w[((oc * CH + ic) * 3 + kh) * 3 + kw];
    }
    __bf16 hi, lo;
    split_bf16(v, hi, lo);
    whi[d * WPACK_PER_D + i] = hi;
    wlo[d * WPACK_PER_D + i] = lo;
  }
}

// ---------------------------------------------------------------------------
// Kernel 2: one block = (batch, dilation-group, oc-tile16, image row).
// 128 threads = 4 waves; wave w computes 16oc x 32px (2 N-subtiles).
// K loop: 9 taps x 2 ic-halves = 18 chunks of K=32; 3 WMMAs per chunk/N-tile.
// ---------------------------------------------------------------------------
__global__ __launch_bounds__(128) void gsd_conv_wmma(
    const float* __restrict__ x,
    const int* __restrict__ counts,
    const int* __restrict__ chan,
    const __bf16* __restrict__ whi,
    const __bf16* __restrict__ wlo,
    float* __restrict__ out) {
  const int d_idx   = blockIdx.z >> 2;   // 0..5
  const int oc_tile = blockIdx.z & 3;    // 0..3
  const int cnt = counts[d_idx];
  if (oc_tile * 16 >= cnt) return;       // uniform early exit (empty tile)
  const int dil = d_idx + 1;
  const int b = blockIdx.y;
  const int h = blockIdx.x;

  extern __shared__ __bf16 smem[];       // [2][3][PXP][IC_PAD] hi then lo
  __bf16* xhi = smem;
  __bf16* xlo = smem + 3 * PXP * IC_PAD;

  const int tid = threadIdx.x;

  // Stage rows h-d, h, h+d incl. zero halo columns. Global reads coalesced
  // along px; LDS layout [row3][px+HALO][ic] (ic contiguous) so B fragments
  // are two unconditional 16B loads per lane (ic +0..7, +16..23).
  for (int i = tid; i < 3 * PXP * CH; i += 128) {
    const int px8  = i % PXP;                   // 0..143 -> px = px8-HALO
    const int row3 = (i / PXP) % 3;
    const int ic   = i / (3 * PXP);
    const int px = px8 - HALO;
    const int r = h + dil * (row3 - 1);
    float v = 0.0f;
    if (r >= 0 && r < HW && px >= 0 && px < HW)
      v = x[(((size_t)b * CH + ic) * HW + r) * HW + px];
    __bf16 hi, lo;
    split_bf16(v, hi, lo);
    const int li = (row3 * PXP + px8) * IC_PAD + ic;
    xhi[li] = hi;
    xlo[li] = lo;
  }
  __syncthreads();

  const int lane = tid & 31;             // wave32
  const int wave = tid >> 5;
  const int half = lane >> 4;            // 0: lanes 0-15, 1: lanes 16-31
  const int l16  = lane & 15;
  const int wb   = wave * 32;            // this wave's pixel base

  v8f acc[2] = {v8f{}, v8f{}};

  // A fragment source: slot row = oc_tile*16 + (lane%16), K-contiguous.
  const size_t slotg = ((size_t)d_idx * CH + oc_tile * 16 + l16) * KTOT;

  for (int tap = 0; tap < 9; ++tap) {
    const int kh = tap / 3, kw = tap % 3;
    const int sx = dil * (kw - 1);       // horizontal shift, in [-6,6]
    const int rowbase = kh * PXP;
#pragma unroll
    for (int cb = 0; cb < 2; ++cb) {
      const int chunk = tap * 2 + cb;
      // ISA 16-bit A layout: lanes<16 hold K+0..7 & K+16..23;
      // lanes>=16 hold K+8..15 & K+24..31.
      const int ka = chunk * 32 + half * 8;
      Frag ahi, alo;
      const uint4* pah = (const uint4*)(whi + slotg + ka);
      const uint4* pal = (const uint4*)(wlo + slotg + ka);
      ahi.q[0] = pah[0]; ahi.q[1] = pah[2];
      alo.q[0] = pal[0]; alo.q[1] = pal[2];
#pragma unroll
      for (int nt = 0; nt < 2; ++nt) {
        // B layout: N = lane%16; lanes<16 hold K 0..7 & 16..23, lanes>=16 K 8..15 & 24..31.
        // Halo-padded LDS -> always in range, zeros outside the image.
        const int px8 = wb + nt * 16 + l16 + sx + HALO;   // 2..141
        const int le = (rowbase + px8) * IC_PAD + cb * 32 + half * 8;
        Frag bhi, blo;
        const uint4* pbh = (const uint4*)(xhi + le);
        const uint4* pbl = (const uint4*)(xlo + le);
        bhi.q[0] = pbh[0]; bhi.q[1] = pbh[2];
        blo.q[0] = pbl[0]; blo.q[1] = pbl[2];
        // D = A*B + C ; three compensated products accumulate in f32.
        acc[nt] = __builtin_amdgcn_wmma_f32_16x16x32_bf16(
            false, ahi.v, false, bhi.v, (short)0, acc[nt], false, false);
        acc[nt] = __builtin_amdgcn_wmma_f32_16x16x32_bf16(
            false, ahi.v, false, blo.v, (short)0, acc[nt], false, false);
        acc[nt] = __builtin_amdgcn_wmma_f32_16x16x32_bf16(
            false, alo.v, false, bhi.v, (short)0, acc[nt], false, false);
      }
    }
  }

  // C/D layout: n = lane%16, m = vgpr + 8*(lane>=16). Scatter real channels.
  const int* chd = chan + d_idx * CH;
#pragma unroll
  for (int nt = 0; nt < 2; ++nt) {
    const int pxb = wb + nt * 16 + l16;
#pragma unroll
    for (int rI = 0; rI < 8; ++rI) {
      const int slot = oc_tile * 16 + rI + 8 * half;
      if (slot < cnt) {
        const int oc = chd[slot];
        out[(((size_t)b * CH + oc) * HW + h) * HW + pxb] = acc[nt][rI];
      }
    }
  }
}

// ---------------------------------------------------------------------------
extern "C" void kernel_launch(void* const* d_in, const int* in_sizes, int n_in,
                              void* d_out, int out_size, void* d_ws, size_t ws_size,
                              hipStream_t stream) {
  const float* x   = (const float*)d_in[0];
  const float* w   = (const float*)d_in[1];
  const int* offs  = (const int*)d_in[2];
  float* out       = (float*)d_out;

  // Workspace layout: counts[16] ints | chan[6][64] ints | (pad to 4KB) |
  // whi[6][64][576] bf16 | wlo[6][64][576] bf16   (~0.87 MB total)
  int* counts = (int*)d_ws;
  int* chan   = counts + 16;
  __bf16* whi = (__bf16*)((char*)d_ws + 4096);
  __bf16* wlo = whi + (size_t)NDIL * WPACK_PER_D;

  gsd_pack_weights<<<NDIL, 256, 0, stream>>>(w, offs, counts, chan, whi, wlo);

  dim3 grid(HW, BATCH, NDIL * 4);        // rows x batch x (dilation, oc-tile)
  const size_t lds = (size_t)2 * 3 * PXP * IC_PAD * sizeof(__bf16); // 124,416 B
  gsd_conv_wmma<<<grid, 128, lds, stream>>>(x, counts, chan, whi, wlo, out);
}
